// NeighborhoodCrossAttention_74766790688938
// MI455X (gfx1250) — matrix operand order repaired and checked by reading, not compiled, not run on hardware
//
#include <hip/hip_runtime.h>
#include <hip/hip_bf16.h>
#include <math.h>

// ---------------------------------------------------------------------------
// NeighborhoodCrossAttention on MI455X (gfx1250, wave32).
// Compute-bound (~39 GFLOP, all tensors L2-resident @192MB, HBM 23.3TB/s is
// irrelevant) -> every GEMM-shaped FLOP goes through v_wmma_f32_16x16x32_bf16.
// Softmax reductions use DPP (quad_perm/row_ror) instead of ds_bpermute.
// ---------------------------------------------------------------------------

typedef __bf16 bf16;
typedef __attribute__((ext_vector_type(8)))  bf16  bf16x8;   // 16 bytes
typedef __attribute__((ext_vector_type(16))) bf16  v16bf;    // 8 VGPRs
typedef __attribute__((ext_vector_type(8)))  float v8f;      // 8 VGPRs

#define Bsz   2
#define NTOK  4096
#define Dm    256
#define NH    8
#define HD    32
#define MROWS (Bsz*NTOK)   // 8192

// ---- WMMA operand loaders -------------------------------------------------
// A-matrix 16x32 bf16 (ISA 7.12.2): lanes 0-15 = rows, K in {g*8..g*8+7} U
// {16+g*8..16+g*8+7} where g = lane>>4.  Two 16B loads per lane.
__device__ __forceinline__ v16bf load_fragA(const bf16* __restrict__ base,
                                            int ld, int lane) {
  const int r = lane & 15;
  const int g = (lane >> 4) & 1;
  const bf16* p = base + (size_t)r * ld;
  union { v16bf v; bf16x8 h[2]; } u;
  u.h[0] = *(const bf16x8*)(p + g * 8);
  u.h[1] = *(const bf16x8*)(p + 16 + g * 8);
  return u.v;
}

// B-matrix 32x16 bf16: lane n holds column n; K in {g*8..g*8+7} U
// {16+g*8..16+g*8+7} where g = lane>>4 (same K order as A).
__device__ __forceinline__ v16bf load_fragB(const bf16* __restrict__ base,
                                            int ld, int lane) {
  const int r = lane & 15;
  const int g = (lane >> 4) & 1;
  const bf16* p = base + (size_t)r * ld + g * 8;
  union { v16bf v; bf16x8 h[2]; } u;
  u.h[0] = *(const bf16x8*)(p);
  u.h[1] = *(const bf16x8*)(p + 16);
  return u.v;
}

__device__ __forceinline__ v8f wmma_bf16(v16bf a, v16bf b, v8f c) {
  return __builtin_amdgcn_wmma_f32_16x16x32_bf16(
      false, a, false, b, (short)0, c, false, false);
}

// ---- DPP 16-lane row reductions (VALU, no LDS traffic) --------------------
template <int CTRL>
__device__ __forceinline__ float dpp_mov(float v) {
  const int s = __builtin_bit_cast(int, v);
  const int d = __builtin_amdgcn_update_dpp(s, s, CTRL, 0xF, 0xF, true);
  return __builtin_bit_cast(float, d);
}
__device__ __forceinline__ float rmax16(float v) {
  v = fmaxf(v, dpp_mov<0xB1>(v));    // quad_perm [1,0,3,2]  (xor 1)
  v = fmaxf(v, dpp_mov<0x4E>(v));    // quad_perm [2,3,0,1]  (xor 2)
  v = fmaxf(v, dpp_mov<0x124>(v));   // row_ror:4
  v = fmaxf(v, dpp_mov<0x128>(v));   // row_ror:8
  return v;
}
__device__ __forceinline__ float rsum16(float v) {
  v += dpp_mov<0xB1>(v);
  v += dpp_mov<0x4E>(v);
  v += dpp_mov<0x124>(v);
  v += dpp_mov<0x128>(v);
  return v;
}

// ---- Kernel 1: fp32 -> bf16 conversions -----------------------------------
__global__ void k_convert(const float* __restrict__ x1,
                          const float* __restrict__ Wq, const float* __restrict__ Wk,
                          const float* __restrict__ Wv, const float* __restrict__ Wo,
                          bf16* __restrict__ x1b,
                          bf16* __restrict__ Wqb, bf16* __restrict__ Wkb,
                          bf16* __restrict__ Wvb, bf16* __restrict__ Wob) {
  int i = blockIdx.x * blockDim.x + threadIdx.x;
  const int NX = MROWS * Dm;
  const int NW = Dm * Dm;
  if (i < NX) { x1b[i] = (bf16)x1[i]; return; }
  i -= NX;
  if (i < NW) { Wqb[i] = (bf16)Wq[i]; return; }
  i -= NW;
  if (i < NW) { Wkb[i] = (bf16)Wk[i]; return; }
  i -= NW;
  if (i < NW) { Wvb[i] = (bf16)Wv[i]; return; }
  i -= NW;
  if (i < NW) { Wob[i] = (bf16)Wo[i]; }
}

// ---- Kernel 2: neighborhood fuse (one wave32 per token) -------------------
__global__ void __launch_bounds__(256) k_fuse(const float* __restrict__ x2,
                                              bf16* __restrict__ x2f) {
  const int lane = threadIdx.x & 31;
  const int w = blockIdx.x * (blockDim.x >> 5) + (threadIdx.x >> 5);
  const int b = w >> 12;
  const int n = w & (NTOK - 1);
  const int row = n >> 6, col = n & 63;
  const float* xb = x2 + (size_t)b * NTOK * Dm;
  const float* cptr = xb + (size_t)n * Dm + lane * 8;
  float c[8];
#pragma unroll
  for (int j = 0; j < 8; ++j) c[j] = cptr[j];

  int idx[10]; bool vld[10];
  idx[0] = n; vld[0] = true;
#pragma unroll
  for (int k = 1; k < 10; ++k) {
    int dr = (k - 1) / 3 - 1, dc = (k - 1) % 3 - 1;
    int rr = row + dr, cc = col + dc;
    bool v = (rr >= 0) && (rr < 64) && (cc >= 0) && (cc < 64);
    vld[k] = v; idx[k] = v ? rr * 64 + cc : 0;
  }

  float sc[10];
#pragma unroll
  for (int k = 0; k < 10; ++k) {
    const float* np = xb + (size_t)idx[k] * Dm + lane * 8;
    float p = 0.f;
#pragma unroll
    for (int j = 0; j < 8; ++j) p += c[j] * np[j];
#pragma unroll
    for (int m = 1; m < 32; m <<= 1) p += __shfl_xor(p, m, 32);
    sc[k] = vld[k] ? p * (1.0f / 16.0f) : -3.0e38f;   // /sqrt(256)
  }
  float mx = sc[0];
#pragma unroll
  for (int k = 1; k < 10; ++k) mx = fmaxf(mx, sc[k]);
  float s = 0.f, wgt[10];
#pragma unroll
  for (int k = 0; k < 10; ++k) { wgt[k] = __expf(sc[k] - mx); s += wgt[k]; }
  const float inv = 1.0f / s;

  float acc[8];
#pragma unroll
  for (int j = 0; j < 8; ++j) acc[j] = wgt[0] * inv * c[j];
#pragma unroll
  for (int k = 1; k < 10; ++k) {
    const float* np = xb + (size_t)idx[k] * Dm + lane * 8;
    const float wk = wgt[k] * inv;
#pragma unroll
    for (int j = 0; j < 8; ++j) acc[j] += wk * np[j];
  }
  bf16* op = x2f + ((size_t)b * NTOK + n) * Dm + lane * 8;
#pragma unroll
  for (int j = 0; j < 8; ++j) op[j] = (bf16)acc[j];
}

// ---- Kernel 3: projection GEMM, X[8192x256] * W^T[256x256] + bias ---------
enum { MODE_HEADS = 0, MODE_VT = 1, MODE_F32OUT = 2 };

template <int MODE>
__global__ void __launch_bounds__(128) k_proj(const bf16* __restrict__ X,
                                              const bf16* __restrict__ W,
                                              const float* __restrict__ bias,
                                              float scale,
                                              void* __restrict__ out) {
  const int lane = threadIdx.x & 31;
  const int tile = blockIdx.x * (blockDim.x >> 5) + (threadIdx.x >> 5);
  const int mt = tile >> 2, ng = tile & 3;
  const int m0 = mt << 4, n0 = ng << 6;          // 16 rows x 64 cols per wave
  const int r = lane & 15, g = (lane >> 4) & 1;

  v8f acc[4];
#pragma unroll
  for (int t = 0; t < 4; ++t) acc[t] = (v8f){0.f,0.f,0.f,0.f,0.f,0.f,0.f,0.f};

#pragma unroll
  for (int kc = 0; kc < Dm; kc += 32) {
    v16bf a = load_fragA(X + (size_t)m0 * Dm + kc, Dm, lane);
#pragma unroll
    for (int t = 0; t < 4; ++t) {
      v16bf bb = load_fragB(W + (size_t)(n0 + 16 * t) * Dm + kc, Dm, lane);
      acc[t] = wmma_bf16(a, bb, acc[t]);
    }
  }

#pragma unroll
  for (int t = 0; t < 4; ++t) {
    const int colo = n0 + 16 * t + r;
    const float bv = bias[colo];
#pragma unroll
    for (int i = 0; i < 8; ++i) {
      const int rowm = m0 + i + 8 * g;
      const float v = (acc[t][i] + bv) * scale;
      if (MODE == MODE_F32OUT) {
        ((float*)out)[(size_t)rowm * Dm + colo] = v;
      } else {
        const int bb = rowm >> 12, n = rowm & (NTOK - 1);
        const int h = colo >> 5, d = colo & 31;
        if (MODE == MODE_HEADS)      // [bh][n][32]
          ((bf16*)out)[(((size_t)(bb * NH + h)) * NTOK + n) * HD + d] = (bf16)v;
        else                         // [bh][32][n]  (V transposed for B-operand)
          ((bf16*)out)[(((size_t)(bb * NH + h)) * HD + d) * NTOK + n] = (bf16)v;
      }
    }
  }
}

// ---- Kernel 4: flash attention, 16 queries/wave, 64 keys/iteration --------
// Q was pre-scaled by (1/sqrt(hd))*log2(e); softmax runs in exp2 domain
// (v_exp_f32 is base-2 natively) which is mathematically identical.
__global__ void __launch_bounds__(128) k_attn(const bf16* __restrict__ Q,
                                              const bf16* __restrict__ K,
                                              const bf16* __restrict__ Vt,
                                              bf16* __restrict__ AO) {
  __shared__ __align__(16) bf16 ldsP[4][16 * 64];   // 2KB per wave: P staging
  const int lane = threadIdx.x & 31;
  const int wid  = threadIdx.x >> 5;
  const int bh   = blockIdx.x >> 6;
  const int q0   = ((blockIdx.x & 63) * 4 + wid) << 4;
  const int r = lane & 15, g = (lane >> 4) & 1;

  const bf16* Qb = Q  + (size_t)bh * NTOK * HD;
  const bf16* Kb = K  + (size_t)bh * NTOK * HD;
  const bf16* Vb = Vt + (size_t)bh * HD * NTOK;
  bf16* myP = ldsP[wid];

  const v16bf qa = load_fragA(Qb + (size_t)q0 * HD, HD, lane);

  const v8f vzero = (v8f){0.f,0.f,0.f,0.f,0.f,0.f,0.f,0.f};
  v8f acc0 = vzero, acc1 = vzero;
  float mrow[8], lrow[8];
#pragma unroll
  for (int i = 0; i < 8; ++i) { mrow[i] = -3.0e38f; lrow[i] = 0.f; }

  for (int k0 = 0; k0 < NTOK; k0 += 64) {
    if (k0 + 64 < NTOK) {   // warm caches for the next chunk
      __builtin_prefetch(Kb + (size_t)(k0 + 64 + r) * HD, 0, 1);
      __builtin_prefetch(Vb + (size_t)r * NTOK + k0 + 64, 0, 1);
    }
    // Issue ALL global fragment loads up front: V-load latency hides under
    // the softmax VALU stretch; K loads batch into clauses.
    const v16bf kb0 = load_fragB(Kb + (size_t)(k0     ) * HD, HD, lane);
    const v16bf kb1 = load_fragB(Kb + (size_t)(k0 + 16) * HD, HD, lane);
    const v16bf kb2 = load_fragB(Kb + (size_t)(k0 + 32) * HD, HD, lane);
    const v16bf kb3 = load_fragB(Kb + (size_t)(k0 + 48) * HD, HD, lane);
    const v16bf vb0 = load_fragB(Vb + k0,                          NTOK, lane);
    const v16bf vb1 = load_fragB(Vb + k0 + 32,                     NTOK, lane);
    const v16bf vb2 = load_fragB(Vb + (size_t)16 * NTOK + k0,      NTOK, lane);
    const v16bf vb3 = load_fragB(Vb + (size_t)16 * NTOK + k0 + 32, NTOK, lane);

    v8f s0 = wmma_bf16(qa, kb0, vzero);
    v8f s1 = wmma_bf16(qa, kb1, vzero);
    v8f s2 = wmma_bf16(qa, kb2, vzero);
    v8f s3 = wmma_bf16(qa, kb3, vzero);

#pragma unroll
    for (int i = 0; i < 8; ++i) {
      const float a0 = s0[i], a1 = s1[i], a2 = s2[i], a3 = s3[i];
      float mx = fmaxf(fmaxf(a0, a1), fmaxf(a2, a3));
      mx = rmax16(mx);                       // row max over 64 keys (DPP)
      const float mn = fmaxf(mrow[i], mx);
      const float cf = __builtin_amdgcn_exp2f(mrow[i] - mn);
      const float e0 = __builtin_amdgcn_exp2f(a0 - mn);
      const float e1 = __builtin_amdgcn_exp2f(a1 - mn);
      const float e2 = __builtin_amdgcn_exp2f(a2 - mn);
      const float e3 = __builtin_amdgcn_exp2f(a3 - mn);
      const float rs = rsum16((e0 + e1) + (e2 + e3));
      lrow[i] = lrow[i] * cf + rs;
      mrow[i] = mn;
      acc0[i] *= cf; acc1[i] *= cf;
      const int rowm = i + 8 * g;            // C-layout -> row-major LDS
      myP[rowm * 64 + r]      = (bf16)e0;
      myP[rowm * 64 + 16 + r] = (bf16)e1;
      myP[rowm * 64 + 32 + r] = (bf16)e2;
      myP[rowm * 64 + 48 + r] = (bf16)e3;
    }
    asm volatile("s_wait_dscnt 0" ::: "memory");   // same-wave DS RAW fence
    const v16bf pa0 = load_fragA(myP,      64, lane);   // keys k0   .. k0+31
    const v16bf pa1 = load_fragA(myP + 32, 64, lane);   // keys k0+32.. k0+63
    acc0 = wmma_bf16(pa0, vb0, acc0);
    acc0 = wmma_bf16(pa1, vb1, acc0);
    acc1 = wmma_bf16(pa0, vb2, acc1);
    acc1 = wmma_bf16(pa1, vb3, acc1);
  }

  const int b = bh >> 3, h = bh & 7;
#pragma unroll
  for (int i = 0; i < 8; ++i) {
    const int rowm = q0 + i + 8 * g;
    const float inv = 1.0f / lrow[i];
    const size_t base = ((size_t)b * NTOK + rowm) * Dm + h * HD;
    AO[base + r]      = (bf16)(acc0[i] * inv);
    AO[base + 16 + r] = (bf16)(acc1[i] * inv);
  }
}

// ---- host-side launch ------------------------------------------------------
extern "C" void kernel_launch(void* const* d_in, const int* in_sizes, int n_in,
                              void* d_out, int out_size, void* d_ws, size_t ws_size,
                              hipStream_t stream) {
  const float* x1 = (const float*)d_in[0];
  const float* x2 = (const float*)d_in[1];
  const float* Wq = (const float*)d_in[2];
  const float* Wk = (const float*)d_in[3];
  const float* Wv = (const float*)d_in[4];
  const float* Wo = (const float*)d_in[5];
  const float* bq = (const float*)d_in[6];
  const float* bk = (const float*)d_in[7];
  const float* bv = (const float*)d_in[8];
  const float* bo = (const float*)d_in[9];

  char* ws = (char*)d_ws;
  bf16* x1b = (bf16*)(ws + 0x000000);            // 4 MB  [8192][256]
  bf16* AO  = x1b;                               // reuse: dead after Q proj
  bf16* x2f = (bf16*)(ws + 0x400000);            // 4 MB  [8192][256]
  bf16* Wqb = (bf16*)(ws + 0x800000);            // 128 KB each
  bf16* Wkb = (bf16*)(ws + 0x820000);
  bf16* Wvb = (bf16*)(ws + 0x840000);
  bf16* Wob = (bf16*)(ws + 0x860000);
  bf16* Qh  = (bf16*)(ws + 0x880000);            // 4 MB  [16][4096][32]
  bf16* Kh  = (bf16*)(ws + 0xC80000);            // 4 MB  [16][4096][32]
  bf16* Vt  = (bf16*)(ws + 0x1080000);           // 4 MB  [16][32][4096]

  // Fold 1/sqrt(hd) and log2(e) into Q so attention softmax is pure exp2.
  const float qscale = 0.17677669529663687f * 1.4426950408889634f;

  k_convert<<<9216, 256, 0, stream>>>(x1, Wq, Wk, Wv, Wo, x1b, Wqb, Wkb, Wvb, Wob);
  k_fuse<<<1024, 256, 0, stream>>>(x2, x2f);
  k_proj<MODE_HEADS><<<512, 128, 0, stream>>>(x1b, Wqb, bq, qscale, Qh);
  k_proj<MODE_HEADS><<<512, 128, 0, stream>>>(x2f, Wkb, bk, 1.0f, Kh);
  k_proj<MODE_VT>   <<<512, 128, 0, stream>>>(x2f, Wvb, bv, 1.0f, Vt);
  k_attn<<<1024, 128, 0, stream>>>(Qh, Kh, Vt, AO);
  k_proj<MODE_F32OUT><<<512, 128, 0, stream>>>(AO, Wob, bo, 1.0f, d_out);
}
